// MainNetwork_76905684402519
// MI455X (gfx1250) — matrix operand order repaired and checked by reading, not compiled, not run on hardware
//
#include <hip/hip_runtime.h>

// ---------------------------------------------------------------------------
// Problem constants (from the reference)
// ---------------------------------------------------------------------------
#define MAP_X   32
#define MAP_Y   24
#define XY      768          // MAP_X * MAP_Y
#define N_ROOMS 48
#define WMAX    8
#define HMAX    8
#define EMB     6
#define CIN0    16           // 9 + 1(ones) + 6(emb)

typedef __attribute__((ext_vector_type(16))) _Float16 v16h;
typedef __attribute__((ext_vector_type(8)))  _Float16 v8h;
typedef __attribute__((ext_vector_type(8)))  float    v8f;

// A-fragment K mapping (matches B's contiguous order):
//   lane L: m = L%16, h = L/16 ; element e -> k = 16*h + e
__host__ __device__ inline int kmapA(int e, int h) {
    return 16 * h + e;
}
// B-fragment order chosen by this code (not AMD's table order; A uses the same): lane n = L%16 ; lanes 0-15 hold K=0..15,
// lanes 16-31 hold K=16..31, K packed sequentially -> k = 16*h + e.
// => a lane's 16 B elements are K-CONTIGUOUS: we exploit this for b128 loads.

// ---------------------------------------------------------------------------
// Weight packing: W[oc][k] (fp32) -> Apack[((kt*2+h)*OC + oc)*16 + e] (f16)
// so each lane's v16h A fragment is one contiguous aligned 32B load.
// ---------------------------------------------------------------------------
__global__ void pack_kernel(const float* __restrict__ W, _Float16* __restrict__ Apack,
                            int OC, int Ktot, int ksteps) {
    int total = ksteps * 2 * OC * 16;
    for (int i = blockIdx.x * blockDim.x + threadIdx.x; i < total;
         i += gridDim.x * blockDim.x) {
        int e    = i & 15;
        int rest = i >> 4;
        int oc   = rest % OC;
        int q    = rest / OC;
        int h    = q & 1;
        int kt   = q >> 1;
        int k    = kt * 32 + kmapA(e, h);
        Apack[i] = (k < Ktot) ? (_Float16)W[oc * Ktot + k] : (_Float16)0.0f;
    }
}

// ---------------------------------------------------------------------------
// encode_map: scatter-add room tiles into the per-sample 16x768 map (LDS fp32
// accumulation via ds_add_f32 atomics), then convert to f16 for the WMMA convs.
// ---------------------------------------------------------------------------
__global__ void encode_kernel(const int* __restrict__ rp,
                              const float* __restrict__ rt,
                              const float* __restrict__ remb,
                              _Float16* __restrict__ X0) {
    __shared__ float smap[CIN0 * XY];   // 48 KB
    const int n   = blockIdx.x;
    const int tid = threadIdx.x;

    for (int i = tid; i < CIN0 * XY; i += blockDim.x)
        smap[i] = (i >= 9 * XY && i < 10 * XY) ? 1.0f : 0.0f;  // ch 9 = ones
    __syncthreads();

    for (int t = tid; t < N_ROOMS * WMAX * HMAX; t += blockDim.x) {
        int r    = t >> 6;
        int cell = t & 63;
        int wx   = cell >> 3;
        int hy   = cell & 7;
        int px   = rp[(n * N_ROOMS + r) * 2 + 0];
        int py   = rp[(n * N_ROOMS + r) * 2 + 1];
        int idx  = (px + wx) * MAP_Y + (py + hy);   // same flat-index (+drop) rule as ref
        if (idx < 0 || idx >= XY) continue;
        float base = rt[((r * 9 + 0) * WMAX + wx) * HMAX + hy];
        #pragma unroll
        for (int c = 0; c < 9; ++c) {
            float v = rt[((r * 9 + c) * WMAX + wx) * HMAX + hy];
            if (v != 0.0f) atomicAdd(&smap[c * XY + idx], v);
        }
        if (base != 0.0f) {
            #pragma unroll
            for (int e = 0; e < EMB; ++e)
                atomicAdd(&smap[(10 + e) * XY + idx], remb[r * EMB + e] * base);
        }
    }
    __syncthreads();

    for (int i = tid; i < CIN0 * XY; i += blockDim.x)
        X0[n * CIN0 * XY + i] = (_Float16)smap[i];
}

// ---------------------------------------------------------------------------
// In-register hoact: lane holds 8 consecutive channels = 2 groups of arity 4.
// out[o] = sum_m ( prod_{j in bits(m)} x_j ) * actW[g][m][o]
// ---------------------------------------------------------------------------
__device__ inline void hoact8(float ch[8], const float* actW, int oc0) {
    #pragma unroll
    for (int gg = 0; gg < 2; ++gg) {
        const int g = (oc0 >> 2) + gg;
        float x[4];
        #pragma unroll
        for (int j = 0; j < 4; ++j) x[j] = ch[gg * 4 + j];
        float f[16];
        f[0] = 1.0f;
        #pragma unroll
        for (int j = 0; j < 4; ++j)
            #pragma unroll
            for (int m = 0; m < (1 << j); ++m)
                f[(1 << j) + m] = f[m] * x[j];
        float out[4] = {0.f, 0.f, 0.f, 0.f};
        #pragma unroll
        for (int m = 0; m < 16; ++m) {
            #pragma unroll
            for (int o = 0; o < 4; ++o)
                out[o] += f[m] * actW[(g * 16 + m) * 4 + o];
        }
        #pragma unroll
        for (int o = 0; o < 4; ++o) ch[gg * 4 + o] = out[o];
    }
}

// ---------------------------------------------------------------------------
// Implicit-GEMM conv + fused bias + hoact.
//   grid = (n, 48) ; tile = 4x4 spatial block (tx in 0..7, ty in 0..5)
//   block = 2*OC threads ; wave w -> output channels [16w, 16w+16)
//   GEMM: C[16 oc x 16 pos] += A[16 oc x 32 k] * B[32 k x 16 pos]
// Stage 1: halo tile -> LDS. Stage 2: im2col -> btile[16][KTOTPAD] so every
// B fragment is K-contiguous => 2x ds_load_b128 per WMMA, immediate offsets.
// FINAL=false -> write f16 [n][OC][768] (next conv input)
// FINAL=true  -> write f32 [n][768][OC] (position-major for the room gather)
// ---------------------------------------------------------------------------
template <int CIN, int OC, int KS, int PAD, bool FINAL>
__global__ void conv_kernel(const _Float16* __restrict__ Xin,
                            const _Float16* __restrict__ Apack,
                            const float* __restrict__ bias,
                            const float* __restrict__ actW,
                            _Float16* __restrict__ XoutH,
                            float* __restrict__ XoutF) {
    constexpr int HW      = 4 + 2 * PAD;       // haloed tile edge
    constexpr int KTOT    = CIN * KS * KS;
    constexpr int KSTEPS  = (KTOT + 31) / 32;
    constexpr int KTOTPAD = KSTEPS * 32;
    constexpr int G       = OC / 4;
    constexpr int NT      = 2 * OC;            // threads per block

    __shared__ _Float16 tile[CIN * HW * HW];
    __shared__ __attribute__((aligned(32))) _Float16 btile[16 * KTOTPAD];
    __shared__ float    sact[G * 16 * 4];

    const int n   = blockIdx.x;
    const int t   = blockIdx.y;
    const int tx  = t / 6;
    const int ty  = t % 6;
    const int tid = threadIdx.x;

    // ---- stage 1: halo tile ----
    for (int e = tid; e < CIN * HW * HW; e += NT) {
        int ic = e / (HW * HW);
        int rr = e % (HW * HW);
        int gx = tx * 4 + (rr / HW) - PAD;
        int gy = ty * 4 + (rr % HW) - PAD;
        _Float16 v = (_Float16)0.0f;
        if (gx >= 0 && gx < MAP_X && gy >= 0 && gy < MAP_Y)
            v = Xin[(n * CIN + ic) * XY + gx * MAP_Y + gy];
        tile[e] = v;
    }
    for (int i = tid; i < G * 64; i += NT) sact[i] = actW[i];
    __syncthreads();

    // ---- stage 2: im2col into btile (no divisions: kx,ky unrolled) ----
    {
        const int nidx = tid & 15;
        const int dxl  = nidx >> 2;
        const int dyl  = nidx & 3;
        const int icb  = tid >> 4;             // 0 .. NT/16-1
        #pragma unroll
        for (int kx = 0; kx < KS; ++kx)
            #pragma unroll
            for (int ky = 0; ky < KS; ++ky)
                for (int ic = icb; ic < CIN; ic += NT / 16)
                    btile[nidx * KTOTPAD + ic * KS * KS + kx * KS + ky] =
                        tile[(ic * HW + dxl + kx) * HW + (dyl + ky)];
        if (KTOT < KTOTPAD) {                  // zero K tail once
            for (int i = tid; i < 16 * (KTOTPAD - KTOT); i += NT) {
                int nn = i % 16;
                int kk = KTOT + i / 16;
                btile[nn * KTOTPAD + kk] = (_Float16)0.0f;
            }
        }
    }
    __syncthreads();

    const int w    = tid >> 5;
    const int lane = tid & 31;
    const int col  = lane & 15;   // A role: m (oc) ; B role: n (spatial)
    const int h    = lane >> 4;

    const _Float16* abase = Apack + ((size_t)h * OC + (w * 16 + col)) * 16;
    const _Float16* bbase = btile + (size_t)col * KTOTPAD + 16 * h;

    v8f acc = {};
    #pragma unroll
    for (int kt = 0; kt < KSTEPS; ++kt) {
        v16h a = *(const v16h*)(abase + (size_t)kt * (2 * OC * 16));
        v16h b = *(const v16h*)(bbase + kt * 32);
        acc = __builtin_amdgcn_wmma_f32_16x16x32_f16(false, a, false, b,
                                                     (short)0, acc, false, false);
    }

    const int oc0 = w * 16 + 8 * h;           // lane holds channels oc0..oc0+7
    float ch[8];
    #pragma unroll
    for (int p = 0; p < 8; ++p) ch[p] = acc[p] + bias[oc0 + p];
    hoact8(ch, sact, oc0);

    const int dx  = col >> 2;
    const int dy  = col & 3;
    const int pos = (tx * 4 + dx) * MAP_Y + (ty * 4 + dy);
    if (!FINAL) {
        #pragma unroll
        for (int p = 0; p < 8; ++p)
            XoutH[(size_t)(n * OC + oc0 + p) * XY + pos] = (_Float16)ch[p];
    } else {
        #pragma unroll
        for (int p = 0; p < 8; ++p)
            XoutF[((size_t)n * XY + pos) * OC + (oc0 + p)] = ch[p];
    }
}

// ---------------------------------------------------------------------------
// decode_map: masked gather-sum of conv3 output (position-major)
//   -> R room-major f16: R[n][48 rooms][128 ch]  (coalesced writes, and the
//      room-GEMM B fragments become contiguous 32B global loads)
// grid = (n, 48 rooms), block = 128 (one thread per channel, coalesced reads)
// ---------------------------------------------------------------------------
__global__ void decode_kernel(const int* __restrict__ rp,
                              const float* __restrict__ rt,
                              const float* __restrict__ X3,
                              _Float16* __restrict__ R0) {
    const int n = blockIdx.x;
    const int r = blockIdx.y;
    const int c = threadIdx.x;
    const int px = rp[(n * N_ROOMS + r) * 2 + 0];
    const int py = rp[(n * N_ROOMS + r) * 2 + 1];
    float sum = 0.0f;
    for (int cell = 0; cell < WMAX * HMAX; ++cell) {
        int wx = cell >> 3, hy = cell & 7;
        float mask = rt[((r * 9 + 0) * WMAX + wx) * HMAX + hy];
        if (mask != 0.0f) {   // mask!=0 implies in-bounds map index
            int idx = (px + wx) * MAP_Y + (py + hy);
            sum += X3[((size_t)n * XY + idx) * 128 + c] * mask;
        }
    }
    R0[((size_t)n * N_ROOMS + r) * 128 + c] = (_Float16)sum;
}

// ---------------------------------------------------------------------------
// Room 1x1 GEMM layer (K=128, room-major activations) + optional fused hoact.
//   grid = (n, 3 room-tiles of 16), block = 2*OCr
//   B fragment: 32B contiguous global load; output: one 16B vector store.
// ---------------------------------------------------------------------------
template <int OCr, bool ACT>
__global__ void room_kernel(const _Float16* __restrict__ Rin,
                            const _Float16* __restrict__ Apack,
                            const float* __restrict__ bias,
                            const float* __restrict__ actW,
                            _Float16* __restrict__ RoutH,
                            float* __restrict__ outF) {
    const int n    = blockIdx.x;
    const int nt   = blockIdx.y;
    const int tid  = threadIdx.x;
    const int w    = tid >> 5;
    const int lane = tid & 31;
    const int col  = lane & 15;
    const int h    = lane >> 4;
    const int room = nt * 16 + col;

    const _Float16* abase = Apack + ((size_t)h * OCr + (w * 16 + col)) * 16;
    const _Float16* bbase = Rin + ((size_t)n * N_ROOMS + room) * 128 + 16 * h;

    v8f acc = {};
    #pragma unroll
    for (int kt = 0; kt < 4; ++kt) {          // K = 128 = 4 * 32
        v16h a = *(const v16h*)(abase + (size_t)kt * (2 * OCr * 16));
        v16h b = *(const v16h*)(bbase + kt * 32);
        acc = __builtin_amdgcn_wmma_f32_16x16x32_f16(false, a, false, b,
                                                     (short)0, acc, false, false);
    }

    const int oc0 = w * 16 + 8 * h;
    float ch[8];
    #pragma unroll
    for (int p = 0; p < 8; ++p) ch[p] = acc[p] + bias[oc0 + p];
    if (ACT) {
        hoact8(ch, actW, oc0);
        v8h hv;
        #pragma unroll
        for (int p = 0; p < 8; ++p) hv[p] = (_Float16)ch[p];
        *(v8h*)(RoutH + ((size_t)n * N_ROOMS + room) * 128 + oc0) = hv;
    } else {
        // final layer: out[n][room][oc] fp32
        #pragma unroll
        for (int p = 0; p < 8; ++p)
            outF[((size_t)n * N_ROOMS + room) * 64 + (oc0 + p)] = ch[p];
    }
}

// ---------------------------------------------------------------------------
// Host launcher
// ---------------------------------------------------------------------------
static inline size_t alignup(size_t x) { return (x + 255) & ~(size_t)255; }

extern "C" void kernel_launch(void* const* d_in, const int* in_sizes, int n_in,
                              void* d_out, int out_size, void* d_ws, size_t ws_size,
                              hipStream_t stream) {
    (void)n_in; (void)out_size; (void)ws_size;
    const int*   rp   = (const int*)  d_in[0];
    const float* rt   = (const float*)d_in[1];
    const float* remb = (const float*)d_in[2];
    const float* mw1  = (const float*)d_in[3];
    const float* mb1  = (const float*)d_in[4];
    const float* ma1  = (const float*)d_in[5];
    const float* mw2  = (const float*)d_in[6];
    const float* mb2  = (const float*)d_in[7];
    const float* ma2  = (const float*)d_in[8];
    const float* mw3  = (const float*)d_in[9];
    const float* mb3  = (const float*)d_in[10];
    const float* ma3  = (const float*)d_in[11];
    const float* rw1  = (const float*)d_in[12];
    const float* rb1  = (const float*)d_in[13];
    const float* ra1  = (const float*)d_in[14];
    const float* rw2  = (const float*)d_in[15];
    const float* rb2  = (const float*)d_in[16];
    const float* ra2  = (const float*)d_in[17];
    const float* rw3  = (const float*)d_in[18];
    const float* rb3  = (const float*)d_in[19];

    const int n = in_sizes[0] / (N_ROOMS * 2);   // batch (256)

    // ---- workspace layout (regions aliased: X0/X2 share, X1/X3 share) ----
    char* ws = (char*)d_ws;
    size_t off = 0;
    _Float16* AP1  = (_Float16*)(ws + off); off = alignup(off + (size_t)13 * 2 * 64  * 16 * 2);
    _Float16* AP2  = (_Float16*)(ws + off); off = alignup(off + (size_t)18 * 2 * 128 * 16 * 2);
    _Float16* AP3  = (_Float16*)(ws + off); off = alignup(off + (size_t)36 * 2 * 128 * 16 * 2);
    _Float16* APR1 = (_Float16*)(ws + off); off = alignup(off + (size_t)4  * 2 * 128 * 16 * 2);
    _Float16* APR2 = (_Float16*)(ws + off); off = alignup(off + (size_t)4  * 2 * 128 * 16 * 2);
    _Float16* APR3 = (_Float16*)(ws + off); off = alignup(off + (size_t)4  * 2 * 64  * 16 * 2);
    // region A: X0 (f16 n*16*768) then reused as X2 (f16 n*128*768)
    char* regA = ws + off; off = alignup(off + (size_t)n * 128 * XY * 2);
    // region B: X1 (f16 n*64*768) then reused as X3 (f32 n*768*128)
    char* regB = ws + off; off = alignup(off + (size_t)n * XY * 128 * 4);
    _Float16* R0 = (_Float16*)(ws + off); off = alignup(off + (size_t)n * N_ROOMS * 128 * 2);
    _Float16* R1 = (_Float16*)(ws + off); off = alignup(off + (size_t)n * N_ROOMS * 128 * 2);

    _Float16* X0 = (_Float16*)regA;
    _Float16* X1 = (_Float16*)regB;
    _Float16* X2 = (_Float16*)regA;
    float*    X3 = (float*)regB;

    // ---- weight packing (tiny) ----
    auto packs = [&](const float* W, _Float16* A, int OC, int Ktot, int ks) {
        int total = ks * 2 * OC * 16;
        pack_kernel<<<(total + 255) / 256, 256, 0, stream>>>(W, A, OC, Ktot, ks);
    };
    packs(mw1, AP1,  64, 16 * 25, 13);
    packs(mw2, AP2, 128, 64 * 9,  18);
    packs(mw3, AP3, 128, 128 * 9, 36);
    packs(rw1, APR1, 128, 128, 4);
    packs(rw2, APR2, 128, 128, 4);
    packs(rw3, APR3, 64,  128, 4);

    // ---- encode ----
    encode_kernel<<<n, 256, 0, stream>>>(rp, rt, remb, X0);

    // ---- conv stack (WMMA) ----
    conv_kernel<16, 64, 5, 2, false><<<dim3(n, 48), 128, 0, stream>>>(X0, AP1, mb1, ma1, X1, nullptr);
    conv_kernel<64, 128, 3, 1, false><<<dim3(n, 48), 256, 0, stream>>>(X1, AP2, mb2, ma2, X2, nullptr);
    conv_kernel<128, 128, 3, 1, true><<<dim3(n, 48), 256, 0, stream>>>(X2, AP3, mb3, ma3, nullptr, X3);

    // ---- decode ----
    decode_kernel<<<dim3(n, N_ROOMS), 128, 0, stream>>>(rp, rt, X3, R0);

    // ---- room stack (WMMA) ----
    room_kernel<128, true ><<<dim3(n, 3), 256, 0, stream>>>(R0, APR1, rb1, ra1, R1, nullptr);
    room_kernel<128, true ><<<dim3(n, 3), 256, 0, stream>>>(R1, APR2, rb2, ra2, R0, nullptr);
    room_kernel<64,  false><<<dim3(n, 3), 128, 0, stream>>>(R0, APR3, rb3, nullptr, nullptr, (float*)d_out);
}
